// RetentiveLayer_53017076302029
// MI455X (gfx1250) — compile-verified, matrix operand order checked by reading, not run on hardware
//
#include <hip/hip_runtime.h>
#include <hip/hip_bf16.h>

// ---------------------------------------------------------------------------
// Problem constants (from reference): B=8, T=8192, D=512, O=512
// ---------------------------------------------------------------------------
#define BB   8
#define TT   8192
#define DD   512
#define OO   512
#define MTOT (BB * TT)          // 65536 rows of the big GEMM

typedef __bf16 v16bf __attribute__((ext_vector_type(16)));
typedef float  v8f   __attribute__((ext_vector_type(8)));

struct U128 { unsigned int a, b, c, d; };
struct U64  { unsigned int a, b; };

static __device__ __forceinline__ unsigned short f2bf(float f) {
    unsigned int u = __float_as_uint(f);
    u += 0x7FFFu + ((u >> 16) & 1u);          // round-to-nearest-even
    return (unsigned short)(u >> 16);
}

static __device__ __forceinline__ float clamp_lam(const float* decay) {
    float l = decay[0];
    return fminf(fmaxf(l, 0.5f), 0.999f);
}

// ---------------------------------------------------------------------------
// Kernel 1: transpose + convert W (f32 [2D, O]) into N-major bf16 W_h^T, W_s^T
// ---------------------------------------------------------------------------
__global__ void prep_w_kernel(const float* __restrict__ W,
                              unsigned short* __restrict__ WtH,
                              unsigned short* __restrict__ WtS) {
    int gid = blockIdx.x * blockDim.x + threadIdx.x;   // 0 .. 262143
    int d = gid >> 9;
    int o = gid & 511;
    WtH[o * DD + d] = f2bf(W[d * OO + o]);
    WtS[o * DD + d] = f2bf(W[(d + DD) * OO + o]);
}

// ---------------------------------------------------------------------------
// Kernel 2: h_avg[t,d] = mean over batch of x  (float4 per thread)
// ---------------------------------------------------------------------------
__global__ void mean_b_kernel(const float* __restrict__ x,
                              float* __restrict__ h_avg) {
    int gid = blockIdx.x * blockDim.x + threadIdx.x;   // 0 .. (T*D/4 - 1)
    const float4* x4 = (const float4*)x;
    float sx = 0.f, sy = 0.f, sz = 0.f, sw = 0.f;
#pragma unroll
    for (int b = 0; b < BB; ++b) {
        float4 v = x4[(size_t)b * (TT * DD / 4) + gid];
        sx += v.x; sy += v.y; sz += v.z; sw += v.w;
    }
    float4 r; r.x = sx * 0.125f; r.y = sy * 0.125f; r.z = sz * 0.125f; r.w = sw * 0.125f;
    ((float4*)h_avg)[gid] = r;
}

// ---------------------------------------------------------------------------
// Blocked EMA scan: 64 chunks of length 128 over T.
// ---------------------------------------------------------------------------
#define NCHUNK 64
#define CLEN   128

// pass 1: local scan within each chunk (parallel over chunk x d)
__global__ void scan1_kernel(const float* __restrict__ h_avg,
                             const float* __restrict__ decay,
                             float* __restrict__ states,
                             float* __restrict__ endv) {
    int tid = threadIdx.x;                       // 0..255
    int d = ((blockIdx.x & 1) << 8) + tid;       // 0..511
    int c = blockIdx.x >> 1;                     // 0..63
    float lam = clamp_lam(decay);
    float om  = 1.0f - lam;
    float s = 0.0f;
    int base = (c * CLEN) * DD + d;
    for (int i = 0; i < CLEN; ++i) {
        float a = h_avg[base + i * DD];
        s = lam * s + om * a;
        states[base + i * DD] = s;
    }
    endv[c * DD + d] = s;
}

// pass 2: serial carry combine over the 64 chunk boundaries
__global__ void scan2_kernel(const float* __restrict__ endv,
                             const float* __restrict__ decay,
                             float* __restrict__ carry) {
    int d = threadIdx.x;                         // 512 threads
    float lam  = clamp_lam(decay);
    float lamL = __powf(lam, (float)CLEN);
    float S = 0.0f;
    for (int c = 0; c < NCHUNK; ++c) {
        carry[c * DD + d] = S;
        S = endv[c * DD + d] + lamL * S;
    }
}

// pass 3: states[t,d] += lam^(i+1) * carry[chunk(t), d]
__global__ void scan3_kernel(float* __restrict__ states,
                             const float* __restrict__ carry,
                             const float* __restrict__ decay) {
    int gid = blockIdx.x * blockDim.x + threadIdx.x;   // 0 .. T*D-1
    int t = gid >> 9;
    int d = gid & 511;
    int c = t >> 7;
    int i = t & (CLEN - 1);
    float lam = clamp_lam(decay);
    float f = __powf(lam, (float)(i + 1));
    states[gid] += f * carry[c * DD + d];
}

// ---------------------------------------------------------------------------
// WMMA GEMM:  out[M,512] = bf16(A[M,512]) @ Bt^T  + aux
//   MAIN=true : aux[(m & (TT-1)) * 512 + n]   (per-t bias from ts)
//   MAIN=false: aux[n]                        (plain bias b)
//   Bt is N-major bf16 [512][512] (i.e. Bt[n][k] = B[k][n]).
//   Workgroup: 256 threads (8 waves), tile 128(M) x 128(N), K-step 32,
//   each wave computes 64x32 via 8 accumulators of v_wmma_f32_16x16x32_bf16.
// ---------------------------------------------------------------------------
#define LDS_STRIDE 40   // halfwords per row: 80B, keeps 16B alignment

static __device__ __forceinline__ void load_tileA(const float* __restrict__ A,
                                                  unsigned short (*dst)[LDS_STRIDE],
                                                  int m0, int k0, int tid) {
#pragma unroll
    for (int it = 0; it < 4; ++it) {
        int chunk = tid + it * 256;              // 0..1023, 4 floats each
        int row = chunk >> 3;                    // 0..127
        int kc  = (chunk & 7) << 2;              // 0,4,..,28
        float4 v = *(const float4*)(A + (size_t)(m0 + row) * DD + k0 + kc);
        unsigned int lo = (unsigned int)f2bf(v.x) | ((unsigned int)f2bf(v.y) << 16);
        unsigned int hi = (unsigned int)f2bf(v.z) | ((unsigned int)f2bf(v.w) << 16);
        U64 pk; pk.a = lo; pk.b = hi;
        *(U64*)&dst[row][kc] = pk;
    }
}

static __device__ __forceinline__ void load_tileB(const unsigned short* __restrict__ Bt,
                                                  unsigned short (*dst)[LDS_STRIDE],
                                                  int n0, int k0, int tid) {
#pragma unroll
    for (int it = 0; it < 2; ++it) {
        int chunk = tid + it * 256;              // 0..511, 8 bf16 each
        int row = chunk >> 2;                    // 0..127
        int kc  = (chunk & 3) << 3;              // 0,8,16,24
        U128 v = *(const U128*)(Bt + (size_t)(n0 + row) * DD + k0 + kc);
        *(U128*)&dst[row][kc] = v;
    }
}

template <bool MAIN>
__global__ __launch_bounds__(256) void gemm_wmma_kernel(
        const float* __restrict__ A,
        const unsigned short* __restrict__ Bt,
        const float* __restrict__ aux,
        float* __restrict__ out) {
    __shared__ unsigned short Alds[2][128][LDS_STRIDE];
    __shared__ unsigned short Blds[2][128][LDS_STRIDE];

    const int tid  = threadIdx.x;
    const int lane = tid & 31;
    const int wave = tid >> 5;                   // 0..7
    const int wm   = wave & 1;                   // 2 waves along M
    const int wn   = wave >> 1;                  // 4 waves along N
    const int m0   = (blockIdx.x >> 2) << 7;     // N/128 == 4 tiles
    const int n0   = (blockIdx.x & 3) << 7;

    v8f acc[4][2];
#pragma unroll
    for (int i = 0; i < 4; ++i)
#pragma unroll
        for (int j = 0; j < 2; ++j)
#pragma unroll
            for (int r = 0; r < 8; ++r) acc[i][j][r] = 0.0f;

    int buf = 0;
    load_tileA(A,  Alds[0], m0, 0, tid);
    load_tileB(Bt, Blds[0], n0, 0, tid);
    __syncthreads();

    const int half8 = (lane >> 4) << 3;          // 0 or 8 (K-half select)
    const int r16   = lane & 15;

    for (int kt = 0; kt < 16; ++kt) {
        if (kt < 15) {
            load_tileA(A,  Alds[buf ^ 1], m0, (kt + 1) * 32, tid);
            load_tileB(Bt, Blds[buf ^ 1], n0, (kt + 1) * 32, tid);
        }
        // pull the next-next A tile toward the caches (global_prefetch_b8,
        // no LOADcnt cost) while the WMMAs below execute
        if (kt < 14 && tid < 128) {
            __builtin_prefetch(A + (size_t)(m0 + tid) * DD + (kt + 2) * 32, 0, 0);
        }

        v16bf aF[4], bF[2];
#pragma unroll
        for (int i = 0; i < 4; ++i) {
            const unsigned short* p = &Alds[buf][wm * 64 + i * 16 + r16][half8];
            union { v16bf v; U128 q[2]; } u;
            u.q[0] = *(const U128*)p;            // K 0..7   (or 8..15)
            u.q[1] = *(const U128*)(p + 16);     // K 16..23 (or 24..31)
            aF[i] = u.v;
        }
#pragma unroll
        for (int j = 0; j < 2; ++j) {
            const unsigned short* p = &Blds[buf][wn * 32 + j * 16 + r16][half8];
            union { v16bf v; U128 q[2]; } u;
            u.q[0] = *(const U128*)p;
            u.q[1] = *(const U128*)(p + 16);
            bF[j] = u.v;
        }

#pragma unroll
        for (int i = 0; i < 4; ++i)
#pragma unroll
            for (int j = 0; j < 2; ++j)
                acc[i][j] = __builtin_amdgcn_wmma_f32_16x16x32_bf16(
                    false, aF[i], false, bF[j], (short)0, acc[i][j], false, false);

        __syncthreads();
        buf ^= 1;
    }

    // epilogue: D-matrix layout -> lanes 0-15: rows r, lanes 16-31: rows r+8.
    // All indexing is 32-bit affine in r (tiles never straddle the T boundary).
    const int mhalf = (lane >> 4) << 3;
#pragma unroll
    for (int i = 0; i < 4; ++i) {
#pragma unroll
        for (int j = 0; j < 2; ++j) {
            const int mBase   = m0 + wm * 64 + i * 16 + mhalf;
            const int n       = n0 + wn * 32 + j * 16 + r16;
            const int outBase = mBase * OO + n;
            if (MAIN) {
                const int auxBase = (mBase & (TT - 1)) * OO + n;
#pragma unroll
                for (int r = 0; r < 8; ++r)
                    out[outBase + r * OO] = acc[i][j][r] + aux[auxBase + r * OO];
            } else {
                const float add = aux[n];
#pragma unroll
                for (int r = 0; r < 8; ++r)
                    out[outBase + r * OO] = acc[i][j][r] + add;
            }
        }
    }
}

// ---------------------------------------------------------------------------
// Launch
// ---------------------------------------------------------------------------
extern "C" void kernel_launch(void* const* d_in, const int* in_sizes, int n_in,
                              void* d_out, int out_size, void* d_ws, size_t ws_size,
                              hipStream_t stream) {
    const float* x     = (const float*)d_in[0];   // (8, 8192, 512)
    const float* W     = (const float*)d_in[1];   // (1024, 512)
    const float* b     = (const float*)d_in[2];   // (512,)
    const float* decay = (const float*)d_in[3];   // scalar
    float* out = (float*)d_out;                   // (8, 8192, 512)

    char* ws = (char*)d_ws;
    float* h_avg          = (float*)(ws + 0);                 // 16 MB
    float* states         = (float*)(ws + (size_t)16777216);  // 16 MB
    float* ts             = (float*)(ws + (size_t)33554432);  // 16 MB  (states@W_s + b)
    float* endv           = (float*)(ws + (size_t)50331648);  // 128 KB
    float* carry          = (float*)(ws + (size_t)50462720);  // 128 KB
    unsigned short* WtH   = (unsigned short*)(ws + (size_t)50593792); // 512 KB
    unsigned short* WtS   = (unsigned short*)(ws + (size_t)51118080); // 512 KB

    // W -> N-major bf16
    prep_w_kernel<<<1024, 256, 0, stream>>>(W, WtH, WtS);
    // mean over batch
    mean_b_kernel<<<(TT * DD / 4) / 256, 256, 0, stream>>>(x, h_avg);
    // blocked EMA scan
    scan1_kernel<<<NCHUNK * 2, 256, 0, stream>>>(h_avg, decay, states, endv);
    scan2_kernel<<<1, 512, 0, stream>>>(endv, decay, carry);
    scan3_kernel<<<(TT * DD) / 256, 256, 0, stream>>>(states, carry, decay);
    // ts = states @ W_s + b          (M=8192  -> 64 M-tiles * 4 N-tiles)
    gemm_wmma_kernel<false><<<256, 256, 0, stream>>>(states, WtS, b, ts);
    // out = x @ W_h + ts[t]          (M=65536 -> 512 M-tiles * 4 N-tiles)
    gemm_wmma_kernel<true><<<2048, 256, 0, stream>>>(x, WtH, ts, out);
}
